// BaseLSTM_4140348473720
// MI455X (gfx1250) — matrix operand order repaired and checked by reading, not compiled, not run on hardware
//
#include <hip/hip_runtime.h>

typedef __attribute__((ext_vector_type(16))) _Float16 v16h;
typedef __attribute__((ext_vector_type(8)))  float    v8f;

#define HS    512   // hidden
#define NF    128   // features
#define BATCH 256
#define TIME  512
#define NOUT  128

// ---------------- helpers ----------------

// A-matrix 16x32 f16 tile (M x K). Per ISA 7.12.2:
// lanes 0-15: m=lane, vec elems [0..7]=K[0..7],  [8..15]=K[16..23]
// lanes 16-31: m=lane-16, vec elems [0..7]=K[8..15], [8..15]=K[24..31]
__device__ __forceinline__ v16h load_a_tile(const _Float16* base, int ld,
                                            int colBase, int lane16, int hsel) {
  const _Float16* p = base + lane16 * ld + colBase + hsel * 8;
  v16h a;
  *((uint4*)&a)     = *(const uint4*)(p);
  *(((uint4*)&a)+1) = *(const uint4*)(p + 16);
  return a;
}

// B-matrix 32x16 f16 tile (K x N), B(k,n) = W[rowBase+n, kBase+k] for row-major W.
// lanes 0-15: n=lane, K=0..7,16..23 ; lanes 16-31: n=lane-16, K=8..15,24..31 (same as A)
__device__ __forceinline__ v16h load_b_tile(const _Float16* p /* W + (rowBase+lane16)*ld + kBase + hsel*8 */) {
  v16h b;
  *((uint4*)&b)     = *(const uint4*)(p);
  *(((uint4*)&b)+1) = *(const uint4*)(p + 16);
  return b;
}

__device__ __forceinline__ float sigmoidf_(float x) {
  return 1.0f / (1.0f + __expf(-x));
}

// ---------------- setup kernels ----------------

__global__ void f32_to_f16_kernel(const float* __restrict__ src,
                                  _Float16* __restrict__ dst, int n) {
  int i = blockIdx.x * blockDim.x + threadIdx.x;
  int stride = gridDim.x * blockDim.x;
  for (; i < n; i += stride) dst[i] = (_Float16)src[i];
}

__global__ void zero_state_kernel(_Float16* __restrict__ h, float* __restrict__ c, int n) {
  int i = blockIdx.x * blockDim.x + threadIdx.x;
  if (i < n) { h[i] = (_Float16)0.0f; c[i] = 0.0f; }
}

// ---------------- LSTM step ----------------
// One wave per 16x16 (batch x hidden) tile; computes all 4 gate tiles.
// grid = 64 blocks x 256 threads (8 waves) -> 512 tiles = 16 Mtiles x 32 Ntiles.
__global__ __launch_bounds__(256)
void lstm_step_kernel(const _Float16* __restrict__ x_h,   // [B,T,F]
                      const _Float16* __restrict__ wi,    // [4H,F]
                      const _Float16* __restrict__ wh,    // [4H,HS]
                      const float* __restrict__ b_i,      // [4H]
                      const float* __restrict__ b_h,      // [4H]
                      const _Float16* __restrict__ h_in,  // [B,HS]
                      _Float16* __restrict__ h_out,       // [B,HS]
                      float* __restrict__ c,              // [B,HS]
                      int t) {
  const int wave   = threadIdx.x >> 5;
  const int lane   = threadIdx.x & 31;
  const int lane16 = lane & 15;
  const int hsel   = lane >> 4;
  const int tile   = blockIdx.x * 8 + wave;
  const int mBase  = (tile & 15) * 16;   // batch rows
  const int nBase  = (tile >> 4) * 16;   // hidden cols

  v8f acc[4] = {v8f{}, v8f{}, v8f{}, v8f{}};

  // ---- recurrent part: gates += h_in @ W_h^T (K = HS) ----
  const _Float16* aBaseH = h_in + mBase * HS;
  #pragma unroll 4
  for (int k = 0; k < HS; k += 32) {
    v16h a = load_a_tile(aBaseH, HS, k, lane16, hsel);
    #pragma unroll
    for (int g = 0; g < 4; ++g) {
      const _Float16* wrow = wh + (size_t)(g * HS + nBase + lane16) * HS + k + hsel * 8;
      v16h b = load_b_tile(wrow);
      acc[g] = __builtin_amdgcn_wmma_f32_16x16x32_f16(false, a, false, b,
                                                      (short)0, acc[g], false, false);
    }
  }

  // ---- fused input projection: gates += x_t @ W_i^T (K = NF) ----
  const _Float16* aBaseX = x_h + (size_t)mBase * (TIME * NF) + (size_t)t * NF;
  #pragma unroll
  for (int k = 0; k < NF; k += 32) {
    v16h a = load_a_tile(aBaseX, TIME * NF, k, lane16, hsel);
    #pragma unroll
    for (int g = 0; g < 4; ++g) {
      const _Float16* wrow = wi + (size_t)(g * HS + nBase + lane16) * NF + k + hsel * 8;
      v16h b = load_b_tile(wrow);
      acc[g] = __builtin_amdgcn_wmma_f32_16x16x32_f16(false, a, false, b,
                                                      (short)0, acc[g], false, false);
    }
  }

  // ---- biases + activations + state update ----
  const int col = nBase + lane16;
  float bias[4];
  #pragma unroll
  for (int g = 0; g < 4; ++g) bias[g] = b_i[g * HS + col] + b_h[g * HS + col];

  #pragma unroll
  for (int r = 0; r < 8; ++r) {
    const int row = mBase + r + hsel * 8;            // C/D layout: M = r + 8*hsel
    const float gi = sigmoidf_(acc[0][r] + bias[0]);
    const float gf = sigmoidf_(acc[1][r] + bias[1]);
    const float gg = tanhf   (acc[2][r] + bias[2]);
    const float go = sigmoidf_(acc[3][r] + bias[3]);
    float cv = c[row * HS + col];
    cv = gf * cv + gi * gg;
    c[row * HS + col] = cv;
    h_out[row * HS + col] = (_Float16)(go * tanhf(cv));
  }
}

// ---------------- final linear: out = h @ W_lin^T + b_lin ----------------
// 16 Mtiles x 8 Ntiles = 128 waves -> 16 blocks x 256 threads
__global__ __launch_bounds__(256)
void lstm_out_kernel(const _Float16* __restrict__ h,    // [B,HS]
                     const _Float16* __restrict__ wl,   // [NOUT,HS]
                     const float* __restrict__ b_lin,   // [NOUT]
                     float* __restrict__ out) {         // [B,NOUT]
  const int wave   = threadIdx.x >> 5;
  const int lane   = threadIdx.x & 31;
  const int lane16 = lane & 15;
  const int hsel   = lane >> 4;
  const int tile   = blockIdx.x * 8 + wave;
  const int mBase  = (tile & 15) * 16;
  const int nBase  = (tile >> 4) * 16;   // 0..7 tiles of 16

  v8f acc = v8f{};
  const _Float16* aBase = h + mBase * HS;
  #pragma unroll 4
  for (int k = 0; k < HS; k += 32) {
    v16h a = load_a_tile(aBase, HS, k, lane16, hsel);
    const _Float16* wrow = wl + (size_t)(nBase + lane16) * HS + k + hsel * 8;
    v16h b = load_b_tile(wrow);
    acc = __builtin_amdgcn_wmma_f32_16x16x32_f16(false, a, false, b,
                                                 (short)0, acc, false, false);
  }
  const int col = nBase + lane16;
  const float bias = b_lin[col];
  #pragma unroll
  for (int r = 0; r < 8; ++r)
    out[(mBase + r + hsel * 8) * NOUT + col] = acc[r] + bias;
}

// ---------------- host ----------------

extern "C" void kernel_launch(void* const* d_in, const int* in_sizes, int n_in,
                              void* d_out, int out_size, void* d_ws, size_t ws_size,
                              hipStream_t stream) {
  const float* x     = (const float*)d_in[0];
  const float* W_i   = (const float*)d_in[1];
  const float* b_i   = (const float*)d_in[2];
  const float* W_h   = (const float*)d_in[3];
  const float* b_h   = (const float*)d_in[4];
  const float* W_lin = (const float*)d_in[5];
  const float* b_lin = (const float*)d_in[6];
  float* out = (float*)d_out;

  char* ws = (char*)d_ws;
  // workspace layout (all 256B-aligned)
  _Float16* x_h  = (_Float16*)(ws);                         // 256*512*128*2 = 32 MiB
  _Float16* wi_h = (_Float16*)(ws + 33554432);              // 2048*128*2
  _Float16* wh_h = (_Float16*)(ws + 34078720);              // 2048*512*2
  _Float16* wl_h = (_Float16*)(ws + 36175872);              // 128*512*2
  _Float16* hA   = (_Float16*)(ws + 36306944);              // 256*512*2
  _Float16* hB   = (_Float16*)(ws + 36569088);              // 256*512*2
  float*    c    = (float*)   (ws + 36831232);              // 256*512*4
  (void)in_sizes; (void)n_in; (void)out_size; (void)ws_size;

  // convert inputs / weights to f16
  f32_to_f16_kernel<<<2048, 256, 0, stream>>>(x,     x_h,  BATCH * TIME * NF);
  f32_to_f16_kernel<<<256,  256, 0, stream>>>(W_i,   wi_h, 4 * HS * NF);
  f32_to_f16_kernel<<<1024, 256, 0, stream>>>(W_h,   wh_h, 4 * HS * HS);
  f32_to_f16_kernel<<<64,   256, 0, stream>>>(W_lin, wl_h, NOUT * HS);

  // h0 = 0, c0 = 0
  zero_state_kernel<<<(BATCH * HS + 255) / 256, 256, 0, stream>>>(hA, c, BATCH * HS);

  // sequential recurrence; h ping-pongs A<->B, ends in hA (t=511 odd: B->A)
  for (int t = 0; t < TIME; ++t) {
    const _Float16* hin  = (t & 1) ? hB : hA;
    _Float16*       hout = (t & 1) ? hA : hB;
    lstm_step_kernel<<<64, 256, 0, stream>>>(x_h, wi_h, wh_h, b_i, b_h, hin, hout, c, t);
  }

  // final projection
  lstm_out_kernel<<<16, 256, 0, stream>>>(hA, wl_h, b_lin, out);
}
